// LongcatFlashMoE_29300266893622
// MI455X (gfx1250) — matrix operand order repaired and checked by reading, not compiled, not run on hardware
//
#include <hip/hip_runtime.h>
#include <hip/hip_bf16.h>
#include <math.h>

// ---------------- problem constants (match reference) ----------------
#define HDIM 2048      // hidden
#define IDIM 1024      // expert ffn hidden
#define NEXP 32        // routed experts
#define NCLS 64        // E + Z router classes
#define TOPK 8
#define TOK  2048      // B*S
#define NSLOT (TOK * TOPK)
#define CAPE 1024      // per-expert capacity
#define RSCALE 1.5f
#define REPS 1e-20f

#define MT_PER_E (CAPE / 16)        // 64 M-tiles per expert
#define KT_H     (HDIM / 32)        // 64 K-tiles over hidden
#define KT_I     (IDIM / 32)        // 32 K-tiles over ffn hidden
#define FRAG_ELEMS 512              // 32 lanes * 16 bf16 per fragment

typedef __attribute__((ext_vector_type(16))) __bf16 bf16x16;
typedef __attribute__((ext_vector_type(8)))  float  floatx8;
typedef __attribute__((ext_vector_type(4)))  int    i32x4;

#if defined(__AMDGCN__) && __has_builtin(__builtin_amdgcn_global_load_async_to_lds_b128) && __has_builtin(__builtin_amdgcn_s_wait_asynccnt)
#define USE_ASYNC 1
#define GLOBAL_AS __attribute__((address_space(1)))
#define LDS_AS    __attribute__((address_space(3)))
#endif

__device__ __forceinline__ float fast_rcp(float x) {
#if defined(__AMDGCN__) && __has_builtin(__builtin_amdgcn_rcpf)
    return __builtin_amdgcn_rcpf(x);
#else
    return 1.0f / x;
#endif
}

__device__ __forceinline__ floatx8 wmma_bf16(bf16x16 a, bf16x16 b, floatx8 c) {
    return __builtin_amdgcn_wmma_f32_16x16x32_bf16(false, a, false, b, (short)0, c, false, false);
}

// =====================================================================
// Fragment-major layouts (everything pre-swizzled for contiguous b128 loads)
//   B frags: [e][n_tile][k_tile][lane 32][elem 16]
//     lane holds column n_tile*16 + (lane&15); elem j -> K = k_tile*32 + 8*(lane>>4) + (j&7) + 16*(j>>3)
//   A frags: [e][m_tile][k_tile][lane 32][elem 16]
//     lane<16: row m, elems 0-7 -> K 0-7,  8-15 -> K 16-23 (of the 32-K tile)
//     lane>=16: row m, elems 0-7 -> K 8-15, 8-15 -> K 24-31
// =====================================================================

// ---------------- kernel 1/2: fp32 -> bf16 + reformat weights into B-fragment layout ----------------
__global__ void reformat_b_kernel(const float* __restrict__ src, __bf16* __restrict__ dst,
                                  int Kdim, int Ndim, int nexp) {
    const int ktiles = Kdim >> 5;
    const int ntiles = Ndim >> 4;
    const size_t fragsPerE = (size_t)ntiles * ktiles;
    const size_t total = (size_t)nexp * fragsPerE * 32;
    const size_t stride = (size_t)gridDim.x * blockDim.x;
    for (size_t tid = (size_t)blockIdx.x * blockDim.x + threadIdx.x; tid < total; tid += stride) {
        const int lane = (int)(tid & 31);
        const size_t f = tid >> 5;
        const int    e  = (int)(f / fragsPerE);
        const size_t fr = f % fragsPerE;
        const int nt = (int)(fr / ktiles);
        const int kt = (int)(fr % ktiles);
        const float* s = src + (size_t)e * Kdim * Ndim
                             + ((size_t)kt * 32 + (size_t)(lane >> 4) * 8) * Ndim
                             + (size_t)nt * 16 + (lane & 15);
        __bf16* d = dst + (f * 32 + lane) * 16;
#pragma unroll
        for (int j = 0; j < 16; ++j) d[j] = (__bf16)s[(size_t)((j & 7) + 16 * (j >> 3)) * Ndim];
    }
}

// ---------------- kernel 3: router (logits -> softmax -> top-8) ----------------
__global__ void router_kernel(const float* __restrict__ x,
                              const float* __restrict__ w_cls,
                              const float* __restrict__ bias,
                              int* __restrict__ flat_ids,   // [NSLOT] expert id, NEXP for zero slots
                              float* __restrict__ w_r,      // [NSLOT] normalized routed weight
                              float* __restrict__ zsum) {   // [TOK] sum of normalized zero weights
    __shared__ float part[4][NCLS];
    __shared__ float sc[NCLS];
    const int t = blockIdx.x;
    const int j = threadIdx.x & (NCLS - 1);
    const int c = threadIdx.x >> 6;
    const int HC = HDIM / 4;
    const float* xr = x + (size_t)t * HDIM + c * HC;
    const float* wr = w_cls + (size_t)j * HDIM + c * HC;
    float s = 0.0f;
    for (int i = 0; i < HC; ++i) s = fmaf(xr[i], wr[i], s);
    part[c][j] = s;
    __syncthreads();
    if (threadIdx.x < NCLS)
        sc[threadIdx.x] = part[0][threadIdx.x] + part[1][threadIdx.x] +
                          part[2][threadIdx.x] + part[3][threadIdx.x];
    __syncthreads();
    if (threadIdx.x == 0) {
        float mx = -3.4e38f;
        for (int i = 0; i < NCLS; ++i) mx = fmaxf(mx, sc[i]);
        float sum = 0.0f;
        for (int i = 0; i < NCLS; ++i) { sc[i] = __expf(sc[i] - mx); sum += sc[i]; }
        const float inv = 1.0f / sum;
        bool taken[NCLS];
        for (int i = 0; i < NCLS; ++i) taken[i] = false;
        int   idx[TOPK];
        float wv[TOPK];
        float wsum = 0.0f;
        for (int k = 0; k < TOPK; ++k) {
            float best = -3.4e38f; int bi = 0;
            for (int i = 0; i < NCLS; ++i) {
                if (!taken[i]) {
                    float v = sc[i] * inv + bias[i];
                    if (v > best) { best = v; bi = i; }
                }
            }
            taken[bi] = true;
            idx[k] = bi;
            wv[k] = sc[bi] * inv;
            wsum += wv[k];
        }
        const float nrm = RSCALE / (wsum + REPS);
        float zs = 0.0f;
        for (int k = 0; k < TOPK; ++k) {
            const float wn = wv[k] * nrm;
            const int n = t * TOPK + k;
            if (idx[k] < NEXP) { flat_ids[n] = idx[k]; w_r[n] = wn; }
            else               { flat_ids[n] = NEXP;   w_r[n] = 0.0f; zs += wn; }
        }
        zsum[t] = zs;
    }
}

// ---------------- kernel 4: deterministic per-expert ranking ----------------
__global__ void rank_kernel(const int* __restrict__ flat_ids,
                            int* __restrict__ pos,
                            int* __restrict__ counts) {
    const int e = threadIdx.x;
    if (e >= NEXP) return;
    int c = 0;
    for (int n = 0; n < NSLOT; ++n) {
        if (flat_ids[n] == e) { pos[n] = c; ++c; }
    }
    counts[e] = c;
}

// ---------------- kernel 5: scatter tokens into A-fragment expert buffers ----------------
// block: 128 threads = 64 k_tiles x 2 lane-halves, each writes 16 contiguous bf16
__global__ void scatter_kernel(const float* __restrict__ x,
                               const int* __restrict__ flat_ids,
                               const int* __restrict__ pos,
                               __bf16* __restrict__ bufA) {
    const int n = blockIdx.x;
    const int e = flat_ids[n];
    if (e >= NEXP) return;
    const int p = pos[n];
    if (p >= CAPE) return;
    const int t = n / TOPK;
    const float* src = x + (size_t)t * HDIM;
    const int mt = p >> 4, r = p & 15;
    const int kt = threadIdx.x >> 1, half = threadIdx.x & 1;
    const int dlane = r + 16 * half;
    __bf16* d = bufA + ((((size_t)e * MT_PER_E + mt) * KT_H + kt) * 32 + dlane) * 16;
#pragma unroll
    for (int j = 0; j < 16; ++j) {
        const int h = 32 * kt + 8 * half + (j & 7) + 16 * (j >> 3);
        d[j] = (__bf16)src[h];
    }
}

// ---------------- kernel 6: grouped GEMM1 (gate/up) + SwiGLU epilogue ----------------
// grid: (IDIM/128, CAPE/16, NEXP); block 256 = 8 waves, one paired gate/up 16x16 tile each.
// A tile staged block-wide into LDS via async copy (double buffered, 8 k_tiles per stage).
__global__ void gemm1_swiglu_kernel(const __bf16* __restrict__ bufA,   // A frags [e][mt][KT_H][32][16]
                                    const __bf16* __restrict__ wguF,   // B frags [e][128][KT_H][32][16]
                                    __bf16* __restrict__ actF,         // A frags [e][mt][KT_I][32][16]
                                    const int* __restrict__ counts) {
    const int e  = blockIdx.z;
    const int mt = blockIdx.y;
    int cnt = counts[e];
    if (cnt > CAPE) cnt = CAPE;
    if (mt * 16 >= cnt) return;
    const int wave = threadIdx.x >> 5;
    const int lane = threadIdx.x & 31;
    const int i0 = (blockIdx.x * 8 + wave) * 16;
    const int ntg = i0 >> 4;
    const __bf16* Abase = bufA + (((size_t)e * MT_PER_E + mt) * KT_H) * FRAG_ELEMS;
    const __bf16* Bg = wguF + (((size_t)e * 128 + ntg) * KT_H) * FRAG_ELEMS;
    const __bf16* Bu = wguF + (((size_t)e * 128 + 64 + ntg) * KT_H) * FRAG_ELEMS;
    floatx8 accg = {};
    floatx8 accu = {};
#ifdef USE_ASYNC
    __shared__ __align__(32) __bf16 lds_a[2][8 * FRAG_ELEMS];   // 2 x 8KB double buffer
    auto stage = [&](int bufi, int chunk) {
        const char* g = (const char*)(Abase + (size_t)chunk * 8 * FRAG_ELEMS);
        char* l = (char*)&lds_a[bufi][0];
        for (int c2 = threadIdx.x; c2 < 512; c2 += 256)
            __builtin_amdgcn_global_load_async_to_lds_b128(
                (GLOBAL_AS i32x4*)(g + (size_t)c2 * 16),
                (LDS_AS i32x4*)(l + (size_t)c2 * 16), 0, 0);
    };
    stage(0, 0);
    for (int ch = 0; ch < 8; ++ch) {
        if (ch + 1 < 8) { stage((ch + 1) & 1, ch + 1); __builtin_amdgcn_s_wait_asynccnt(2); }
        else            { __builtin_amdgcn_s_wait_asynccnt(0); }
        __syncthreads();
        const bf16x16* As = (const bf16x16*)&lds_a[ch & 1][0];
        for (int kt8 = 0; kt8 < 8; ++kt8) {
            const int kt = ch * 8 + kt8;
            bf16x16 a = As[kt8 * 32 + lane];
            const bf16x16* bgp = (const bf16x16*)(Bg + (size_t)kt * FRAG_ELEMS);
            const bf16x16* bup = (const bf16x16*)(Bu + (size_t)kt * FRAG_ELEMS);
            accg = wmma_bf16(a, bgp[lane], accg);
            accu = wmma_bf16(a, bup[lane], accu);
        }
        __syncthreads();
    }
#else
    for (int kt = 0; kt < KT_H; ++kt) {
        const bf16x16* ap  = (const bf16x16*)(Abase + (size_t)kt * FRAG_ELEMS);
        const bf16x16* bgp = (const bf16x16*)(Bg + (size_t)kt * FRAG_ELEMS);
        const bf16x16* bup = (const bf16x16*)(Bu + (size_t)kt * FRAG_ELEMS);
        bf16x16 a = ap[lane];
        accg = wmma_bf16(a, bgp[lane], accg);
        accu = wmma_bf16(a, bup[lane], accu);
    }
#endif
    // epilogue: act = silu(g)*u, written straight into GEMM2's A-fragment layout
    const int col = lane & 15;
    const int i = i0 + col;
    const int kt2 = i >> 5, kl = i & 31;
    const int half = (kl >> 3) & 1;
    const int j = (kl & 7) + 8 * (kl >> 4);
    const int rbase = (lane >> 4) * 8;
    __bf16* dbase = actF + ((((size_t)e * MT_PER_E + mt) * KT_I + kt2) * 32) * 16 + j;
#pragma unroll
    for (int r = 0; r < 8; ++r) {
        const int dlane = (rbase + r) + 16 * half;
        const float g = accg[r];
        const float u = accu[r];
        const float sig = fast_rcp(1.0f + __expf(-g));
        dbase[(size_t)dlane * 16] = (__bf16)(g * sig * u);
    }
}

// ---------------- kernel 7: grouped GEMM2 (down projection, fp32 out) ----------------
// grid: (HDIM/128, CAPE/16, NEXP); block 256
__global__ void gemm2_kernel(const __bf16* __restrict__ actF,   // A frags [e][mt][KT_I][32][16]
                             const __bf16* __restrict__ wdnF,   // B frags [e][128][KT_I][32][16]
                             float* __restrict__ y,             // [NEXP, CAPE, HDIM] row-major
                             const int* __restrict__ counts) {
    const int e  = blockIdx.z;
    const int mt = blockIdx.y;
    int cnt = counts[e];
    if (cnt > CAPE) cnt = CAPE;
    if (mt * 16 >= cnt) return;
    const int wave = threadIdx.x >> 5;
    const int lane = threadIdx.x & 31;
    const int h0 = (blockIdx.x * 8 + wave) * 16;
    const int nt = h0 >> 4;
    const __bf16* Abase = actF + (((size_t)e * MT_PER_E + mt) * KT_I) * FRAG_ELEMS;
    const __bf16* B = wdnF + (((size_t)e * 128 + nt) * KT_I) * FRAG_ELEMS;
    floatx8 acc = {};
#ifdef USE_ASYNC
    __shared__ __align__(32) __bf16 lds_a[2][8 * FRAG_ELEMS];
    auto stage = [&](int bufi, int chunk) {
        const char* g = (const char*)(Abase + (size_t)chunk * 8 * FRAG_ELEMS);
        char* l = (char*)&lds_a[bufi][0];
        for (int c2 = threadIdx.x; c2 < 512; c2 += 256)
            __builtin_amdgcn_global_load_async_to_lds_b128(
                (GLOBAL_AS i32x4*)(g + (size_t)c2 * 16),
                (LDS_AS i32x4*)(l + (size_t)c2 * 16), 0, 0);
    };
    stage(0, 0);
    for (int ch = 0; ch < 4; ++ch) {
        if (ch + 1 < 4) { stage((ch + 1) & 1, ch + 1); __builtin_amdgcn_s_wait_asynccnt(2); }
        else            { __builtin_amdgcn_s_wait_asynccnt(0); }
        __syncthreads();
        const bf16x16* As = (const bf16x16*)&lds_a[ch & 1][0];
        for (int kt8 = 0; kt8 < 8; ++kt8) {
            const int kt = ch * 8 + kt8;
            bf16x16 a = As[kt8 * 32 + lane];
            const bf16x16* bp = (const bf16x16*)(B + (size_t)kt * FRAG_ELEMS);
            acc = wmma_bf16(a, bp[lane], acc);
        }
        __syncthreads();
    }
#else
    for (int kt = 0; kt < KT_I; ++kt) {
        const bf16x16* ap = (const bf16x16*)(Abase + (size_t)kt * FRAG_ELEMS);
        const bf16x16* bp = (const bf16x16*)(B + (size_t)kt * FRAG_ELEMS);
        acc = wmma_bf16(ap[lane], bp[lane], acc);
    }
#endif
    float* C = y + ((size_t)e * CAPE + mt * 16) * HDIM + h0 + (lane & 15);
    const int rbase = (lane >> 4) * 8;
#pragma unroll
    for (int r = 0; r < 8; ++r) C[(size_t)(rbase + r) * HDIM] = acc[r];
}

// ---------------- kernel 8: gather + weighted combine + zero-expert scale ----------------
__global__ void combine_kernel(const float* __restrict__ y,
                               const int* __restrict__ flat_ids,
                               const int* __restrict__ pos,
                               const float* __restrict__ w_r,
                               const float* __restrict__ zsum,
                               float* __restrict__ out) {
    const int t = blockIdx.x;
    const float zscale = 1.0f + zsum[t];
    for (int h = threadIdx.x; h < HDIM; h += blockDim.x) {
        float acc = 0.0f;
#pragma unroll
        for (int k = 0; k < TOPK; ++k) {
            const int n = t * TOPK + k;
            const int e = flat_ids[n];
            if (e < NEXP) {
                const int p = pos[n];
                if (p < CAPE)
                    acc = fmaf(w_r[n], y[((size_t)e * CAPE + p) * HDIM + h], acc);
            }
        }
        out[(size_t)t * HDIM + h] = acc * zscale;
    }
}

// ---------------- launcher ----------------
extern "C" void kernel_launch(void* const* d_in, const int* in_sizes, int n_in,
                              void* d_out, int out_size, void* d_ws, size_t ws_size,
                              hipStream_t stream) {
    const float* x         = (const float*)d_in[0];   // [TOK, HDIM]
    const float* w_cls     = (const float*)d_in[1];   // [NCLS, HDIM]
    const float* bias      = (const float*)d_in[2];   // [NCLS]
    const float* w_gate_up = (const float*)d_in[3];   // [NEXP, HDIM, 2*IDIM]
    const float* w_down    = (const float*)d_in[4];   // [NEXP, IDIM, HDIM]
    float* out = (float*)d_out;

    char* base = (char*)d_ws;
    size_t off = 0;
    auto take = [&](size_t bytes) -> void* {
        void* r = base + off;
        off += (bytes + 255) & ~(size_t)255;
        return r;
    };
    const size_t n_wgu = (size_t)NEXP * HDIM * 2 * IDIM;
    const size_t n_wdn = (size_t)NEXP * IDIM * HDIM;
    __bf16* wgu_bf = (__bf16*)take(n_wgu * sizeof(__bf16));
    __bf16* wdn_bf = (__bf16*)take(n_wdn * sizeof(__bf16));
    __bf16* buf_bf = (__bf16*)take((size_t)NEXP * CAPE * HDIM * sizeof(__bf16));
    __bf16* act_bf = (__bf16*)take((size_t)NEXP * CAPE * IDIM * sizeof(__bf16));
    float*  ybuf   = (float*)take((size_t)NEXP * CAPE * HDIM * sizeof(float));
    int*    ids    = (int*)take(NSLOT * sizeof(int));
    int*    pos    = (int*)take(NSLOT * sizeof(int));
    float*  wr     = (float*)take(NSLOT * sizeof(float));
    float*  zs     = (float*)take(TOK * sizeof(float));
    int*    cnts   = (int*)take(NEXP * sizeof(int));

    // 1-2) weights -> bf16 fragment-major layouts
    reformat_b_kernel<<<8192, 256, 0, stream>>>(w_gate_up, wgu_bf, HDIM, 2 * IDIM, NEXP);
    reformat_b_kernel<<<8192, 256, 0, stream>>>(w_down, wdn_bf, IDIM, HDIM, NEXP);
    // 3) router
    router_kernel<<<TOK, 256, 0, stream>>>(x, w_cls, bias, ids, wr, zs);
    // 4) deterministic ranking + counts
    rank_kernel<<<1, 64, 0, stream>>>(ids, pos, cnts);
    // 5) dispatch into A-fragment expert buffers
    scatter_kernel<<<NSLOT, 128, 0, stream>>>(x, ids, pos, buf_bf);
    // 6) gate/up GEMM + SwiGLU
    {
        dim3 grid(IDIM / 128, CAPE / 16, NEXP);
        gemm1_swiglu_kernel<<<grid, 256, 0, stream>>>(buf_bf, wgu_bf, act_bf, cnts);
    }
    // 7) down projection GEMM
    {
        dim3 grid(HDIM / 128, CAPE / 16, NEXP);
        gemm2_kernel<<<grid, 256, 0, stream>>>(act_bf, wdn_bf, ybuf, cnts);
    }
    // 8) combine
    combine_kernel<<<TOK, 256, 0, stream>>>(ybuf, ids, pos, wr, zs, out);
}
